// SwinTransformerBlock_32530082300578
// MI455X (gfx1250) — compile-verified, matrix operand order checked by reading, not run on hardware
//
#include <hip/hip_runtime.h>
#include <hip/hip_bf16.h>
#include <cmath>

// ---------------------------------------------------------------------------
// Types / helpers
// ---------------------------------------------------------------------------
typedef __attribute__((ext_vector_type(16))) _Float16 v16h;
typedef __attribute__((ext_vector_type(8)))  float    v8f;

#define DEVINL __device__ __forceinline__

union FragU { v16h v; uint4 q[2]; };

// ---------------------------------------------------------------------------
// CDNA5 async global->LDS copy (ASYNCcnt-tracked DMA), with sync fallback.
// Builtin signature (probe-confirmed via diagnostic): 
//   void __builtin_amdgcn_global_load_async_to_lds_b128(
//        int4 addrspace(1)* src, int4 addrspace(3)* dst, imm offset, imm cpol)
// ---------------------------------------------------------------------------
#if __has_builtin(__builtin_amdgcn_global_load_async_to_lds_b128) && \
    __has_builtin(__builtin_amdgcn_s_wait_asynccnt)
#define HAVE_ASYNC_LDS 1
#else
#define HAVE_ASYNC_LDS 0
#endif

typedef int aint4 __attribute__((vector_size(16)));
typedef __attribute__((address_space(1))) aint4 g_aint4;   // global int4
typedef __attribute__((address_space(3))) aint4 l_aint4;   // LDS int4

DEVINL void async_copy16(const _Float16* gsrc, _Float16* ldst) {
#if HAVE_ASYNC_LDS
    __builtin_amdgcn_global_load_async_to_lds_b128(
        (g_aint4*)gsrc, (l_aint4*)ldst, 0, 0);
#else
    *reinterpret_cast<uint4*>(ldst) = *reinterpret_cast<const uint4*>(gsrc);
#endif
}

template<int N> DEVINL void async_wait() {
#if HAVE_ASYNC_LDS
    __builtin_amdgcn_s_wait_asynccnt(N);
#endif
}

// Load a 16x32 f16 WMMA fragment (A-layout; B uses the same per-lane pattern
// when the matrix is stored [16 rows][K contiguous]).
// Per ISA 7.12.2: lane l (r = l&15, g = l>>4) holds halves
//   e=0..7  -> K = g*8 + e ;  e=8..15 -> K = 16 + g*8 + (e-8)
// i.e. two contiguous 16-byte chunks at offsets g*8 and 16+g*8 halves.
DEVINL v16h load_frag(const _Float16* base, int strideHalves) {
    const int lane = threadIdx.x & 31;
    const _Float16* p = base + (size_t)(lane & 15) * strideHalves + ((lane >> 4) * 8);
    FragU f;
    f.q[0] = *reinterpret_cast<const uint4*>(p);
    f.q[1] = *reinterpret_cast<const uint4*>(p + 16);
    return f.v;
}

DEVINL v8f wmma_f16(v16h a, v16h b, v8f c) {
    return __builtin_amdgcn_wmma_f32_16x16x32_f16(
        /*neg_a=*/false, a, /*neg_b=*/false, b,
        /*c_mod=*/(short)0, c, /*reuse_a=*/false, /*reuse_b=*/false);
}

DEVINL float gelu_exact(float x) {
    return 0.5f * x * (1.0f + erff(x * 0.70710678118654752f));
}

// ---------------------------------------------------------------------------
// Constants
// ---------------------------------------------------------------------------
#define C_DIM   256
#define HEADS   8
#define HDIM    32
#define WS      8
#define SS      4
#define NWIN    64          // tokens per window
#define HW      64          // image side
#define M_TOK   65536       // 16 * 64 * 64
#define ATT_SCALE 0.17677669529663687f   // 32^-0.5

// ---------------------------------------------------------------------------
// Weight transpose + f32 -> f16:  w[K][N] -> wt[N][K]
// ---------------------------------------------------------------------------
__global__ void wtrans_kernel(const float* __restrict__ w, _Float16* __restrict__ wt,
                              int K, int N) {
    int idx = blockIdx.x * 256 + threadIdx.x;
    if (idx < K * N) {
        int n = idx / K;
        int k = idx - n * K;
        wt[idx] = (_Float16)w[(size_t)k * N + n];
    }
}

// ---------------------------------------------------------------------------
// LN1 + cyclic shift (-SS,-SS) + window partition.  One wave per output row.
// ---------------------------------------------------------------------------
__global__ __launch_bounds__(256)
void ln1_window_kernel(const float* __restrict__ x,
                       const float* __restrict__ g, const float* __restrict__ b,
                       _Float16* __restrict__ xw) {
    const int r    = blockIdx.x * 8 + (threadIdx.x >> 5);
    const int lane = threadIdx.x & 31;

    const int bi    = r >> 12;          // / 4096
    const int win   = (r >> 6) & 63;
    const int inner = r & 63;
    const int i  = (win >> 3) * 8 + (inner >> 3);
    const int j  = (win & 7)  * 8 + (inner & 7);
    const int is = (i + SS) & 63;
    const int js = (j + SS) & 63;
    const size_t src = ((size_t)bi * 4096 + is * 64 + js) * C_DIM;

    float4 a0 = *reinterpret_cast<const float4*>(&x[src + lane * 8]);
    float4 a1 = *reinterpret_cast<const float4*>(&x[src + lane * 8 + 4]);
    float s  = a0.x + a0.y + a0.z + a0.w + a1.x + a1.y + a1.z + a1.w;
    float s2 = a0.x*a0.x + a0.y*a0.y + a0.z*a0.z + a0.w*a0.w
             + a1.x*a1.x + a1.y*a1.y + a1.z*a1.z + a1.w*a1.w;
    #pragma unroll
    for (int m = 16; m >= 1; m >>= 1) {
        s  += __shfl_xor(s,  m, 32);
        s2 += __shfl_xor(s2, m, 32);
    }
    const float mean = s * (1.0f / C_DIM);
    const float var  = s2 * (1.0f / C_DIM) - mean * mean;
    const float rstd = rsqrtf(var + 1e-5f);

    FragU o;
    const float in[8] = {a0.x, a0.y, a0.z, a0.w, a1.x, a1.y, a1.z, a1.w};
    #pragma unroll
    for (int e = 0; e < 8; ++e) {
        int c = lane * 8 + e;
        o.v[e] = (_Float16)((in[e] - mean) * rstd * g[c] + b[c]);
    }
    *reinterpret_cast<uint4*>(&xw[(size_t)r * C_DIM + lane * 8]) = o.q[0];
}

// ---------------------------------------------------------------------------
// Unshift + residual + LN2:  y = x + unshift(projout); h = LN2(y) as f16.
// ---------------------------------------------------------------------------
__global__ __launch_bounds__(256)
void ln2_unshift_kernel(const float* __restrict__ x,
                        const float* __restrict__ projout,
                        const float* __restrict__ g, const float* __restrict__ b,
                        _Float16* __restrict__ h) {
    const int t    = blockIdx.x * 8 + (threadIdx.x >> 5);
    const int lane = threadIdx.x & 31;

    const int bi = t >> 12;
    const int l  = t & 4095;
    const int i  = l >> 6;
    const int j  = l & 63;
    const int is = (i + HW - SS) & 63;   // reverse roll
    const int js = (j + HW - SS) & 63;
    const int winrow = ((is >> 3) * 8 + (js >> 3)) * 64 + (is & 7) * 8 + (js & 7);
    const size_t psrc = ((size_t)bi * 4096 + winrow) * C_DIM;
    const size_t xsrc = (size_t)t * C_DIM;

    float y[8];
    float s = 0.f, s2 = 0.f;
    #pragma unroll
    for (int e = 0; e < 8; ++e) {
        int c = lane * 8 + e;
        float v = x[xsrc + c] + projout[psrc + c];
        y[e] = v; s += v; s2 += v * v;
    }
    #pragma unroll
    for (int m = 16; m >= 1; m >>= 1) {
        s  += __shfl_xor(s,  m, 32);
        s2 += __shfl_xor(s2, m, 32);
    }
    const float mean = s * (1.0f / C_DIM);
    const float var  = s2 * (1.0f / C_DIM) - mean * mean;
    const float rstd = rsqrtf(var + 1e-5f);

    FragU o;
    #pragma unroll
    for (int e = 0; e < 8; ++e) {
        int c = lane * 8 + e;
        o.v[e] = (_Float16)((y[e] - mean) * rstd * g[c] + b[c]);
    }
    *reinterpret_cast<uint4*>(&h[xsrc + lane * 8]) = o.q[0];
}

// ---------------------------------------------------------------------------
// Generic WMMA GEMM:  out[M][N] = A_f16[M][K] @ Wt_f16[N][K]^T + bias
// Block: 256 threads (8 waves), tile 128x128, K-step 32.
// Double-buffered LDS pipeline: tile k+1 streams in via async global->LDS DMA
// (ASYNCcnt) while WMMAs consume tile k.  4 async b128 per thread per tile,
// so s_wait_asynccnt<=4 after issuing tile k+1 means tile k has landed.
// EPI: 0 = f32 out, 1 = f16 out, 2 = f16 out + exact GELU, 3 = f32 out + residual
// ---------------------------------------------------------------------------
template<int EPI>
__global__ __launch_bounds__(256)
void gemm_wmma_kernel(const _Float16* __restrict__ A,
                      const _Float16* __restrict__ Wt,
                      const float* __restrict__ bias,
                      const float* __restrict__ residual,
                      void* __restrict__ out,
                      int M, int N, int K) {
    constexpr int LS = 40;                    // LDS stride in halves (pad)
    __shared__ _Float16 sA[2][128 * LS];
    __shared__ _Float16 sB[2][128 * LS];

    const int m0 = blockIdx.y * 128;
    const int n0 = blockIdx.x * 128;
    const int t  = threadIdx.x;
    const int wv = t >> 5;
    const int wm = wv & 3;                    // 0..3 -> row group of 32
    const int wn = wv >> 2;                   // 0..1 -> col group of 64

    v8f acc[2][4];
    #pragma unroll
    for (int i = 0; i < 2; ++i)
        #pragma unroll
        for (int j = 0; j < 4; ++j)
            acc[i][j] = (v8f){0.f,0.f,0.f,0.f,0.f,0.f,0.f,0.f};

    // Issue one 128x32 tile of A and Wt into LDS buffer `buf` (4 x b128/thread).
    auto issue_tile = [&](int k0, int buf) {
        #pragma unroll
        for (int c = 0; c < 2; ++c) {
            int cid = t + c * 256;
            int row = cid >> 2;
            int ch  = (cid & 3) * 8;
            async_copy16(&A [(size_t)(m0 + row) * K + k0 + ch], &sA[buf][row * LS + ch]);
            async_copy16(&Wt[(size_t)(n0 + row) * K + k0 + ch], &sB[buf][row * LS + ch]);
        }
    };

    const int nk = K >> 5;
    issue_tile(0, 0);
    for (int kt = 0; kt < nk; ++kt) {
        const int cur = kt & 1;
        if (kt + 1 < nk) {
            issue_tile((kt + 1) << 5, (kt + 1) & 1);
            async_wait<4>();                  // tile kt has landed (own lane's part)
        } else {
            async_wait<0>();
        }
        __syncthreads();                      // publish all waves' stores

        v16h af[2], bf[4];
        #pragma unroll
        for (int i = 0; i < 2; ++i) af[i] = load_frag(&sA[cur][(wm * 32 + i * 16) * LS], LS);
        #pragma unroll
        for (int j = 0; j < 4; ++j) bf[j] = load_frag(&sB[cur][(wn * 64 + j * 16) * LS], LS);
        #pragma unroll
        for (int i = 0; i < 2; ++i)
            #pragma unroll
            for (int j = 0; j < 4; ++j)
                acc[i][j] = wmma_f16(af[i], bf[j], acc[i][j]);
        __syncthreads();                      // safe to overwrite this buffer next
    }

    // Epilogue. C/D layout: vgpr r -> row r (+8 for lanes 16..31), lane%16 -> col.
    const int lane = t & 31;
    const int lr   = lane & 15;
    const int lhi  = lane >> 4;
    #pragma unroll
    for (int i = 0; i < 2; ++i) {
        #pragma unroll
        for (int j = 0; j < 4; ++j) {
            const int col = n0 + wn * 64 + j * 16 + lr;
            const float bv = bias[col];
            #pragma unroll
            for (int r = 0; r < 8; ++r) {
                const int row = m0 + wm * 32 + i * 16 + r + 8 * lhi;
                const size_t idx = (size_t)row * N + col;
                float v = acc[i][j][r] + bv;
                if (EPI == 2) v = gelu_exact(v);
                if (EPI == 3) v += residual[idx];
                if (EPI == 0 || EPI == 3) reinterpret_cast<float*>(out)[idx] = v;
                else                      reinterpret_cast<_Float16*>(out)[idx] = (_Float16)v;
            }
        }
    }
}

// ---------------------------------------------------------------------------
// Windowed attention: one block (128 threads / 4 waves) per (window, head).
// qkv f16 layout: [65536 win-order tokens][768], col = which*256 + head*32 + d.
// out f16: [65536][256], col = head*32 + d.
// ---------------------------------------------------------------------------
DEVINL int region_class(int t) { return t < (HW - WS) ? 0 : (t < (HW - SS) ? 1 : 2); }

__global__ __launch_bounds__(128)
void attn_kernel(const _Float16* __restrict__ qkv,
                 const float* __restrict__ rpb,     // [225][8]
                 _Float16* __restrict__ out) {
    __shared__ float    S[64 * 66];
    __shared__ _Float16 P[64 * 72];
    __shared__ _Float16 VT[32 * 72];

    const int head = blockIdx.x & 7;
    const int widx = blockIdx.x >> 3;        // 0..1023 window (= b*64 + win)
    const int win  = widx & 63;
    const int t    = threadIdx.x;
    const int wv   = t >> 5;
    const int lane = t & 31;
    const int lr   = lane & 15;
    const int lhi  = lane >> 4;

    const _Float16* base = qkv + (size_t)widx * NWIN * 768 + head * HDIM;

    // ---- S = q @ k^T  (each wave: 16 rows x 64 cols, K = 32) ----
    v16h qf = load_frag(base + (size_t)(wv * 16) * 768, 768);
    v8f s[4];
    #pragma unroll
    for (int j = 0; j < 4; ++j) {
        v16h kf = load_frag(base + 256 + (size_t)(j * 16) * 768, 768);
        v8f z = (v8f){0.f,0.f,0.f,0.f,0.f,0.f,0.f,0.f};
        s[j] = wmma_f16(qf, kf, z);
    }
    #pragma unroll
    for (int j = 0; j < 4; ++j)
        #pragma unroll
        for (int r = 0; r < 8; ++r)
            S[(wv * 16 + r + 8 * lhi) * 66 + j * 16 + lr] = s[j][r];

    // ---- stage V transposed: VT[c][m] = v[m][c] ----
    {
        const int m  = t >> 1;
        const int c0 = (t & 1) * 16;
        FragU u;
        u.q[0] = *reinterpret_cast<const uint4*>(base + 512 + (size_t)m * 768 + c0);
        u.q[1] = *reinterpret_cast<const uint4*>(base + 512 + (size_t)m * 768 + c0 + 8);
        #pragma unroll
        for (int e = 0; e < 16; ++e) VT[(c0 + e) * 72 + m] = u.v[e];
    }
    __syncthreads();

    // ---- softmax rows (scale + rel-pos bias + shift mask) ----
    if (t < 64) {
        const int n  = t;
        const int ny = n >> 3, nx = n & 7;
        const int iw = (win >> 3) * 8 + ny, jw = (win & 7) * 8 + nx;
        const int rid_n = 3 * region_class(iw) + region_class(jw);

        float mx = -3.0e38f;
        for (int m = 0; m < 64; ++m) {
            const int my = m >> 3, mxx = m & 7;
            const int bidx = ((nx - mxx) + 7) * 15 + ((ny - my) + 7);
            const int im = (win >> 3) * 8 + my, jm = (win & 7) * 8 + mxx;
            const int rid_m = 3 * region_class(im) + region_class(jm);
            const float msk = (rid_m == rid_n) ? -100.0f : 0.0f;  // faithful to source
            float v = S[n * 66 + m] * ATT_SCALE + rpb[bidx * 8 + head] + msk;
            S[n * 66 + m] = v;
            mx = fmaxf(mx, v);
        }
        float sum = 0.f;
        for (int m = 0; m < 64; ++m) {
            float e = expf(S[n * 66 + m] - mx);
            sum += e;
            P[n * 72 + m] = (_Float16)e;
        }
        const float inv = 1.0f / sum;
        for (int m = 0; m < 64; ++m)
            P[n * 72 + m] = (_Float16)((float)P[n * 72 + m] * inv);
    }
    __syncthreads();

    // ---- out = P @ V  (each wave: 16 rows x 32 cols, K = 64) ----
    v8f o[2];
    #pragma unroll
    for (int fj = 0; fj < 2; ++fj) o[fj] = (v8f){0.f,0.f,0.f,0.f,0.f,0.f,0.f,0.f};
    #pragma unroll
    for (int kk = 0; kk < 2; ++kk) {
        v16h pf = load_frag(&P[(wv * 16) * 72 + kk * 32], 72);
        #pragma unroll
        for (int fj = 0; fj < 2; ++fj) {
            v16h vf = load_frag(&VT[(fj * 16) * 72 + kk * 32], 72);
            o[fj] = wmma_f16(pf, vf, o[fj]);
        }
    }
    #pragma unroll
    for (int fj = 0; fj < 2; ++fj)
        #pragma unroll
        for (int r = 0; r < 8; ++r) {
            const int row = wv * 16 + r + 8 * lhi;
            const int col = head * HDIM + fj * 16 + lr;
            out[((size_t)widx * NWIN + row) * C_DIM + col] = (_Float16)o[fj][r];
        }
}

// ---------------------------------------------------------------------------
// Host launch
// ---------------------------------------------------------------------------
extern "C" void kernel_launch(void* const* d_in, const int* in_sizes, int n_in,
                              void* d_out, int out_size, void* d_ws, size_t ws_size,
                              hipStream_t stream) {
    const float* x       = (const float*)d_in[0];
    const float* n1g     = (const float*)d_in[1];
    const float* n1b     = (const float*)d_in[2];
    const float* qkv_w   = (const float*)d_in[3];
    const float* qkv_b   = (const float*)d_in[4];
    const float* rpb     = (const float*)d_in[5];
    const float* proj_w  = (const float*)d_in[6];
    const float* proj_b  = (const float*)d_in[7];
    const float* n2g     = (const float*)d_in[8];
    const float* n2b     = (const float*)d_in[9];
    const float* fc1_w   = (const float*)d_in[10];
    const float* fc1_b   = (const float*)d_in[11];
    const float* fc2_w   = (const float*)d_in[12];
    const float* fc2_b   = (const float*)d_in[13];

    // Workspace layout (256B aligned).
    char* ws = (char*)d_ws;
    size_t off = 0;
    auto alloc = [&](size_t bytes) -> void* {
        void* p = ws + off;
        off += (bytes + 255) & ~(size_t)255;
        return p;
    };
    _Float16* wqkv_t = (_Float16*)alloc((size_t)768  * 256  * 2);
    _Float16* wproj_t= (_Float16*)alloc((size_t)256  * 256  * 2);
    _Float16* wfc1_t = (_Float16*)alloc((size_t)1024 * 256  * 2);
    _Float16* wfc2_t = (_Float16*)alloc((size_t)256  * 1024 * 2);
    _Float16* buf1   = (_Float16*)alloc((size_t)M_TOK * 256 * 2);   // xw / attn_out / h
    char*     buf2   = (char*)    alloc((size_t)M_TOK * 1024 * 2);  // qkv / projout / fc1out

    _Float16* qkv16   = (_Float16*)buf2;
    float*    projout = (float*)buf2;
    _Float16* fc1out  = (_Float16*)buf2;

    // 1) weights -> f16, transposed to [out][in]
    wtrans_kernel<<<(256 * 768  + 255) / 256, 256, 0, stream>>>(qkv_w,  wqkv_t, 256,  768);
    wtrans_kernel<<<(256 * 256  + 255) / 256, 256, 0, stream>>>(proj_w, wproj_t,256,  256);
    wtrans_kernel<<<(256 * 1024 + 255) / 256, 256, 0, stream>>>(fc1_w,  wfc1_t, 256,  1024);
    wtrans_kernel<<<(1024* 256  + 255) / 256, 256, 0, stream>>>(fc2_w,  wfc2_t, 1024, 256);

    // 2) LN1 + shift + window partition
    ln1_window_kernel<<<M_TOK / 8, 256, 0, stream>>>(x, n1g, n1b, buf1);

    // 3) QKV GEMM: [65536,256] x [768,256]^T -> f16
    gemm_wmma_kernel<1><<<dim3(768 / 128, M_TOK / 128), 256, 0, stream>>>(
        buf1, wqkv_t, qkv_b, nullptr, qkv16, M_TOK, 768, 256);

    // 4) windowed attention -> attn_out f16 (buf1)
    attn_kernel<<<1024 * HEADS, 128, 0, stream>>>(qkv16, rpb, buf1);

    // 5) proj GEMM: -> f32 window-order (buf2)
    gemm_wmma_kernel<0><<<dim3(256 / 128, M_TOK / 128), 256, 0, stream>>>(
        buf1, wproj_t, proj_b, nullptr, projout, M_TOK, 256, 256);

    // 6) unshift + residual + LN2 -> h f16 (buf1)
    ln2_unshift_kernel<<<M_TOK / 8, 256, 0, stream>>>(x, projout, n2g, n2b, buf1);

    // 7) FC1 + GELU -> f16 (buf2)
    gemm_wmma_kernel<2><<<dim3(1024 / 128, M_TOK / 128), 256, 0, stream>>>(
        buf1, wfc1_t, fc1_b, nullptr, fc1out, M_TOK, 1024, 256);

    // 8) FC2 + bias + original shortcut -> d_out f32
    gemm_wmma_kernel<3><<<dim3(256 / 128, M_TOK / 128), 256, 0, stream>>>(
        fc1out, wfc2_t, fc2_b, x, d_out, M_TOK, 256, 1024);
}
